// NeuroManifoldBlock_24026047054330
// MI455X (gfx1250) — matrix operand order repaired and checked by reading, not compiled, not run on hardware
//
#include <hip/hip_runtime.h>
#include <cstddef>

// ---------------------------------------------------------------------------
// CDNA5 (gfx1250) NeuroManifold block:
//   x = sdr
//   x += FHN-attention(LN1(x))      (Q/K/V/O projections + softmax attention)
//   x += SwiGLU(LN2(x))             (gate/up/down projections)
// GEMMs: v_wmma_f32_16x16x32_bf16 with Tensor-Data-Mover (TDM) double-buffered
// LDS staging (tensor_load_to_lds + s_wait_tensorcnt). Weights transposed at
// convert time; all LDS traffic in hot loops is 128-bit.
// ---------------------------------------------------------------------------

typedef __attribute__((ext_vector_type(16))) __bf16 v16bf;
typedef __attribute__((ext_vector_type(8)))  float  v8f;
typedef unsigned int u32x4 __attribute__((ext_vector_type(4)));
typedef int          i32x8 __attribute__((ext_vector_type(8)));
typedef int          i32x4 __attribute__((ext_vector_type(4)));

union Frag16 { v16bf v; unsigned short u[16]; uint4 q[2]; };
union FragC  { v8f   v; float          f[8];  };
union U4S8   { uint4 q; unsigned short s[8];  };

#if defined(__HIP_DEVICE_COMPILE__) && __has_builtin(__builtin_amdgcn_tensor_load_to_lds)
#define USE_TDM 1
#else
#define USE_TDM 0
#endif

__device__ __forceinline__ unsigned short f32_to_bf16(float f) {
  unsigned int u = __float_as_uint(f);
  u += 0x7FFFu + ((u >> 16) & 1u);      // round-to-nearest-even
  return (unsigned short)(u >> 16);
}
__device__ __forceinline__ float bf16_to_f32(unsigned short h) {
  return __uint_as_float(((unsigned int)h) << 16);
}

#if USE_TDM
// 2D tile DMA: global (bf16 elements, row stride `stride0` elems) -> LDS.
// LDS rows padded: pad_interval code 3 = 16 DWORDs (64B = 32 bf16 = tile row),
// pad_amount code 3 = 4 DWORDs (16B) -> 80B LDS row pitch ([..][40] arrays).
// rem0/rem1 = remaining tensor extent from the tile origin (HW zero-fills OOB).
__device__ __forceinline__ void tdm_load_2d(unsigned int lds_byte_addr,
                                            const void* gaddr,
                                            unsigned int rem0, unsigned int rem1,
                                            unsigned int tile0, unsigned int tile1,
                                            unsigned int stride0) {
  const unsigned long long ga = (unsigned long long)gaddr;
  u32x4 g0 = { 1u,                                   // count=1, user-mode
               lds_byte_addr,                        // lds_addr (bytes)
               (unsigned int)ga,                     // global_addr[31:0]
               (unsigned int)(ga >> 32) | 0x80000000u }; // ga[56:32] | type=2
  i32x8 g1 = { (int)((1u << 16) |                    // data_size = 2 bytes
                     (1u << 20) |                    // pad_enable
                     (3u << 22) |                    // pad_interval: 16 DWORDs
                     (3u << 25)),                    // pad_amount : 4 DWORDs
               (int)((rem0 & 0xFFFFu) << 16),        // tensor_dim0 lo
               (int)((rem0 >> 16) | ((rem1 & 0xFFFFu) << 16)),
               (int)((rem1 >> 16) | (tile0 << 16)),  // dim1 hi | tile_dim0
               (int)tile1,                           // tile_dim1 (tile_dim2=0)
               (int)stride0,                         // tensor_dim0_stride lo
               0, 0 };
  i32x4 z4 = { 0, 0, 0, 0 };
  i32x8 z8 = { 0, 0, 0, 0, 0, 0, 0, 0 };
  __builtin_amdgcn_tensor_load_to_lds(g0, g1, z4, z4, z8, 0);
}
__device__ __forceinline__ unsigned int lds_addr_of(const void* p) {
  return (unsigned int)(unsigned long long)p;   // LDS aperture: low 32 bits
}
#endif

// ---------------------------------------------------------------------------
// Convert + transpose weights: out[N][Kp] (bf16) = in[K][N] (f32).
// Rows k in [K, Kp) are zero-filled so GEMMs can run unguarded vector loads.
// ---------------------------------------------------------------------------
__global__ __launch_bounds__(256)
void cvt_transpose_bf16(const float* __restrict__ in,
                        unsigned short* __restrict__ out,
                        int K, int N, int Kp) {
  __shared__ float tile[32][33];
  const int k0 = blockIdx.y * 32;
  const int n0 = blockIdx.x * 32;
  const int tx = threadIdx.x & 31;
  const int ty = threadIdx.x >> 5;          // 0..7
#pragma unroll
  for (int i = 0; i < 32; i += 8) {
    const int k = k0 + ty + i, n = n0 + tx;
    tile[ty + i][tx] = (k < K && n < N) ? in[(size_t)k * N + n] : 0.f;
  }
  __syncthreads();
#pragma unroll
  for (int i = 0; i < 32; i += 8) {
    const int n = n0 + ty + i, k = k0 + tx;
    if (n < N && k < Kp)
      out[(size_t)n * Kp + k] = f32_to_bf16(tile[tx][ty + i]);
  }
}

// ---------------------------------------------------------------------------
// LayerNorm over last dim (E), output bf16. One block per row.
// ---------------------------------------------------------------------------
__global__ __launch_bounds__(256)
void ln_bf16(const float* __restrict__ x, const float* __restrict__ g,
             const float* __restrict__ b, unsigned short* __restrict__ out, int E) {
  const int row = blockIdx.x;
  const float* xr = x + (size_t)row * E;
  __shared__ float s1[256], s2[256];
  float lsum = 0.f, lsq = 0.f;
  for (int i = threadIdx.x; i < E; i += 256) {
    float v = xr[i]; lsum += v; lsq += v * v;
  }
  s1[threadIdx.x] = lsum; s2[threadIdx.x] = lsq;
  __syncthreads();
  for (int s = 128; s > 0; s >>= 1) {
    if (threadIdx.x < s) { s1[threadIdx.x] += s1[threadIdx.x + s];
                           s2[threadIdx.x] += s2[threadIdx.x + s]; }
    __syncthreads();
  }
  const float mu   = s1[0] / (float)E;
  const float var  = s2[0] / (float)E - mu * mu;
  const float rstd = rsqrtf(var + 1e-5f);
  for (int i = threadIdx.x; i < E; i += 256) {
    float v = (xr[i] - mu) * rstd * g[i] + b[i];
    out[(size_t)row * E + i] = f32_to_bf16(v);
  }
}

// ---------------------------------------------------------------------------
// Tiled bf16 WMMA GEMM:  C[M,N] = A[M,K] * Bt[N,K]^T   (Bt is transposed B)
// lda/ldb multiples of 8, >= roundup32(K), zero padded.
// Block = 256 threads = 8 waves; block tile 64x128; wave tile 32x32.
// TDM path: wave0 DMAs A tile, wave1 DMAs B tile, double-buffered LDS;
// s_wait_tensorcnt(1) overlaps next-tile DMA with WMMA on current tile.
// ---------------------------------------------------------------------------
__global__ __launch_bounds__(256)
void gemm_bf16_wmma(const unsigned short* __restrict__ A,
                    const unsigned short* __restrict__ Bt,
                    void* __restrict__ Cout,
                    int M, int N, int K, int lda, int ldb, int out_bf16) {
#if USE_TDM
  __shared__ __align__(16) unsigned short sA [2][64][40];
  __shared__ __align__(16) unsigned short sBt[2][128][40];
#else
  __shared__ __align__(16) unsigned short sA [1][64][40];
  __shared__ __align__(16) unsigned short sBt[1][128][40];
#endif

  const int tid  = threadIdx.x;
  const int lane = tid & 31;
  const int wid  = tid >> 5;
  const int wr   = wid >> 2;            // 0..1 : 32-row strip
  const int wc   = wid & 3;             // 0..3 : 32-col strip
  const int m16  = lane & 15;
  const int hi   = lane >> 4;

  const int bm = blockIdx.y * 64;
  const int bn = blockIdx.x * 128;

  FragC acc[2][2];
#pragma unroll
  for (int i = 0; i < 2; ++i)
#pragma unroll
    for (int j = 0; j < 2; ++j)
#pragma unroll
      for (int r = 0; r < 8; ++r) acc[i][j].f[r] = 0.f;

  const int nIter = (K + 31) >> 5;

#if USE_TDM
  // prologue: DMA tile 0 into buffer 0
  if (wid == 0) {
    tdm_load_2d(lds_addr_of(&sA[0][0][0]), A + (size_t)bm * lda,
                (unsigned)lda, (unsigned)(M - bm), 32u, 64u, (unsigned)lda);
  } else if (wid == 1) {
    tdm_load_2d(lds_addr_of(&sBt[0][0][0]), Bt + (size_t)bn * ldb,
                (unsigned)ldb, (unsigned)(N - bn), 32u, 128u, (unsigned)ldb);
  }
  for (int it = 0; it < nIter; ++it) {
    const int k0  = it * 32;
    const int buf = it & 1;
    const bool more = (it + 1) < nIter;
    if (wid == 0) {
      if (more) {
        tdm_load_2d(lds_addr_of(&sA[buf ^ 1][0][0]),
                    A + (size_t)bm * lda + k0 + 32,
                    (unsigned)(lda - k0 - 32), (unsigned)(M - bm),
                    32u, 64u, (unsigned)lda);
        __builtin_amdgcn_s_wait_tensorcnt(1);   // tile `it` complete
      } else {
        __builtin_amdgcn_s_wait_tensorcnt(0);
      }
    } else if (wid == 1) {
      if (more) {
        tdm_load_2d(lds_addr_of(&sBt[buf ^ 1][0][0]),
                    Bt + (size_t)bn * ldb + k0 + 32,
                    (unsigned)(ldb - k0 - 32), (unsigned)(N - bn),
                    32u, 128u, (unsigned)ldb);
        __builtin_amdgcn_s_wait_tensorcnt(1);
      } else {
        __builtin_amdgcn_s_wait_tensorcnt(0);
      }
    }
    __syncthreads();                    // tile `it` visible to all waves
#else
  const int ar = tid >> 2;              // 0..63
  const int ac = (tid & 3) * 8;         // 0,8,16,24
  for (int it = 0; it < nIter; ++it) {
    const int k0  = it * 32;
    const int buf = 0;
    __syncthreads();                    // previous iteration's reads done
    {
      const unsigned short* Ap = A + (size_t)(bm + ar) * lda + k0 + ac;
      *(uint4*)&sA[0][ar][ac] = *(const uint4*)Ap;
      if (k0 + 32 < K) __builtin_prefetch(Ap + 32, 0, 1);
    }
#pragma unroll
    for (int i = 0; i < 2; ++i) {
      const int idx = tid * 2 + i;
      const int n   = idx >> 2;
      const int c8  = (idx & 3) * 8;
      uint4 val = make_uint4(0u, 0u, 0u, 0u);
      if (bn + n < N)
        val = *(const uint4*)&Bt[(size_t)(bn + n) * ldb + k0 + c8];
      *(uint4*)&sBt[0][n][c8] = val;
    }
    __syncthreads();
#endif

    // ---- fragments (all ds_load_b128) + WMMA ----
    Frag16 af[2];
#pragma unroll
    for (int i = 0; i < 2; ++i) {
      const int rb = wr * 32 + i * 16 + m16;
      af[i].q[0] = *(const uint4*)&sA[buf][rb][hi * 8];
      af[i].q[1] = *(const uint4*)&sA[buf][rb][16 + hi * 8];
    }
    Frag16 bf2[2];
#pragma unroll
    for (int j = 0; j < 2; ++j) {
      const unsigned short* bp = &sBt[buf][wc * 32 + j * 16 + m16][hi * 8];
      bf2[j].q[0] = *(const uint4*)bp;
      bf2[j].q[1] = *(const uint4*)(bp + 16);
    }
#pragma unroll
    for (int i = 0; i < 2; ++i)
#pragma unroll
      for (int j = 0; j < 2; ++j)
        acc[i][j].v = __builtin_amdgcn_wmma_f32_16x16x32_bf16(
            false, af[i].v, false, bf2[j].v, (short)0, acc[i][j].v,
            false, false);
#if USE_TDM
    __syncthreads();                    // reads done before buf is re-DMAed
#endif
  }

  // ---- epilogue ----
#pragma unroll
  for (int i = 0; i < 2; ++i) {
#pragma unroll
    for (int j = 0; j < 2; ++j) {
      const int gcol = bn + wc * 32 + j * 16 + m16;
      if (gcol < N) {
#pragma unroll
        for (int r = 0; r < 8; ++r) {
          const int grow = bm + wr * 32 + i * 16 + r + 8 * hi;
          if (out_bf16)
            ((unsigned short*)Cout)[(size_t)grow * N + gcol] =
                f32_to_bf16(acc[i][j].f[r]);
          else
            ((float*)Cout)[(size_t)grow * N + gcol] = acc[i][j].f[r];
        }
      }
    }
  }
}

// ---------------------------------------------------------------------------
// Flash-style attention, one block per (64 q-rows, b, h); 4 waves x 16 rows.
// Q/K/V are bf16 [B,T,E], head h at columns h*D .. h*D+127 (D=128).
// K staged row-major; V staged transposed (sVt[d][key]) so P*V fragments are
// contiguous b128 reads. Output ctx is f32 [B,T,E].
// ---------------------------------------------------------------------------
__global__ __launch_bounds__(128)
void attn_wmma(const unsigned short* __restrict__ Qg,
               const unsigned short* __restrict__ Kg,
               const unsigned short* __restrict__ Vg,
               float* __restrict__ ctx,
               int T, int E, int H, int D) {
  __shared__ __align__(16) unsigned short sK [32][136];   // 272B rows
  __shared__ __align__(16) unsigned short sVt[128][40];   // 80B rows
  __shared__ __align__(16) unsigned short sP [4][16][40];

  const int tid  = threadIdx.x;
  const int lane = tid & 31;
  const int wid  = tid >> 5;
  const int m16  = lane & 15;
  const int hi   = lane >> 4;

  const int bh = blockIdx.y;
  const int b  = bh / H;
  const int h  = bh % H;
  const int qbase = blockIdx.x * 64 + wid * 16;

  const size_t headOff = (size_t)b * T * E + (size_t)h * D;
  const unsigned short* Qp = Qg + headOff;
  const unsigned short* Kp = Kg + headOff;
  const unsigned short* Vp = Vg + headOff;

  const float scale = rsqrtf((float)D);

  Frag16 qa[4];
#pragma unroll
  for (int c = 0; c < 4; ++c) {
    const unsigned short* qp = Qp + (size_t)(qbase + m16) * E + c * 32;
    qa[c].q[0] = *(const uint4*)(qp + hi * 8);
    qa[c].q[1] = *(const uint4*)(qp + 16 + hi * 8);
  }

  FragC o[8];
#pragma unroll
  for (int d = 0; d < 8; ++d)
#pragma unroll
    for (int r = 0; r < 8; ++r) o[d].f[r] = 0.f;

  float mst[8], lst[8];
#pragma unroll
  for (int r = 0; r < 8; ++r) { mst[r] = -1e30f; lst[r] = 0.f; }

  const int nkt = T / 32;
  for (int kt = 0; kt < nkt; ++kt) {
    const int keyb = kt * 32;
    __syncthreads();
#pragma unroll
    for (int i = 0; i < 4; ++i) {
      const int idx = tid * 4 + i;
      const int row = idx >> 4;               // key 0..31
      const int c8  = (idx & 15) * 8;         // d 0..120
      *(uint4*)&sK[row][c8] = *(const uint4*)&Kp[(size_t)(keyb + row) * E + c8];
      U4S8 vv;
      vv.q = *(const uint4*)&Vp[(size_t)(keyb + row) * E + c8];
#pragma unroll
      for (int j = 0; j < 8; ++j) sVt[c8 + j][row] = vv.s[j];
    }
    __syncthreads();

    // ---- S = Q K^T (16 x 32) ----
    FragC s[2];
#pragma unroll
    for (int nt = 0; nt < 2; ++nt)
#pragma unroll
      for (int r = 0; r < 8; ++r) s[nt].f[r] = 0.f;
#pragma unroll
    for (int nt = 0; nt < 2; ++nt) {
      const int keyrow = nt * 16 + m16;
#pragma unroll
      for (int c = 0; c < 4; ++c) {
        Frag16 kf;
        const unsigned short* kp = &sK[keyrow][c * 32 + hi * 8];
        kf.q[0] = *(const uint4*)kp;
        kf.q[1] = *(const uint4*)(kp + 16);
        s[nt].v = __builtin_amdgcn_wmma_f32_16x16x32_bf16(
            false, qa[c].v, false, kf.v, (short)0, s[nt].v, false, false);
      }
    }

    // ---- online softmax per row ----
#pragma unroll
    for (int r = 0; r < 8; ++r) {
      float s0 = s[0].f[r] * scale;
      float s1 = s[1].f[r] * scale;
      float mx = fmaxf(s0, s1);
#pragma unroll
      for (int msk = 8; msk > 0; msk >>= 1)
        mx = fmaxf(mx, __shfl_xor(mx, msk, 16));
      const float mnew  = fmaxf(mst[r], mx);
      const float alpha = __expf(mst[r] - mnew);
      mst[r] = mnew;
      const float p0 = __expf(s0 - mnew);
      const float p1 = __expf(s1 - mnew);
      float rs = p0 + p1;
#pragma unroll
      for (int msk = 8; msk > 0; msk >>= 1)
        rs += __shfl_xor(rs, msk, 16);
      lst[r] = lst[r] * alpha + rs;
#pragma unroll
      for (int d = 0; d < 8; ++d) o[d].f[r] *= alpha;
      sP[wid][r + 8 * hi][m16]      = f32_to_bf16(p0);
      sP[wid][r + 8 * hi][16 + m16] = f32_to_bf16(p1);
    }

    Frag16 pf;
    pf.q[0] = *(const uint4*)&sP[wid][m16][hi * 8];
    pf.q[1] = *(const uint4*)&sP[wid][m16][16 + hi * 8];

    // ---- O += P V ----
#pragma unroll
    for (int dt = 0; dt < 8; ++dt) {
      Frag16 vf;
      const unsigned short* vp = &sVt[dt * 16 + m16][hi * 8];
      vf.q[0] = *(const uint4*)vp;
      vf.q[1] = *(const uint4*)(vp + 16);
      o[dt].v = __builtin_amdgcn_wmma_f32_16x16x32_bf16(
          false, pf.v, false, vf.v, (short)0, o[dt].v, false, false);
    }
  }

  float* ctxp = ctx + headOff;
#pragma unroll
  for (int r = 0; r < 8; ++r) {
    const float inv = 1.0f / lst[r];
    const int row = qbase + r + 8 * hi;
#pragma unroll
    for (int dt = 0; dt < 8; ++dt)
      ctxp[(size_t)row * E + dt * 16 + m16] = o[dt].f[r] * inv;
  }
}

// ---------------------------------------------------------------------------
// FHN dynamics (2 Euler steps, v0=w0=0) on ctx; emit v-state as bf16.
// ---------------------------------------------------------------------------
__global__ __launch_bounds__(256)
void fhn_bf16(const float* __restrict__ ctx, unsigned short* __restrict__ vout,
              long long n) {
  long long i = (long long)blockIdx.x * 256 + threadIdx.x;
  if (i >= n) return;
  const float I = ctx[i];
  float v = 0.f, w = 0.f;
#pragma unroll
  for (int s = 0; s < 2; ++s) {
    v += 0.1f * (v - v * v * v * (1.f / 3.f) - w + I);
    w += (0.1f / 12.5f) * (v + 0.7f - 0.8f * w);
  }
  vout[i] = f32_to_bf16(v);
}

// ---------------------------------------------------------------------------
// h[M][Kp] = silu(gate) * up (bf16), zero-filled in the K-pad columns.
// ---------------------------------------------------------------------------
__global__ __launch_bounds__(256)
void silu_mul_pad_bf16(const unsigned short* __restrict__ gate,
                       const unsigned short* __restrict__ up,
                       unsigned short* __restrict__ h,
                       int HID, int Kp, long long n) {
  long long i = (long long)blockIdx.x * 256 + threadIdx.x;
  if (i >= n) return;
  const int row = (int)(i / Kp);
  const int col = (int)(i % Kp);
  float r = 0.f;
  if (col < HID) {
    const float g = bf16_to_f32(gate[(size_t)row * HID + col]);
    const float u = bf16_to_f32(up[(size_t)row * HID + col]);
    r = (g / (1.f + __expf(-g))) * u;
  }
  h[i] = f32_to_bf16(r);
}

// ---------------------------------------------------------------------------
// out = a + b (f32)
// ---------------------------------------------------------------------------
__global__ __launch_bounds__(256)
void add_f32(const float* __restrict__ a, const float* __restrict__ b,
             float* __restrict__ out, long long n) {
  long long i = (long long)blockIdx.x * 256 + threadIdx.x;
  if (i < n) out[i] = a[i] + b[i];
}

// ---------------------------------------------------------------------------
// Host-side orchestration
// ---------------------------------------------------------------------------
static inline int blks(long long n) { return (int)((n + 255) / 256); }

extern "C" void kernel_launch(void* const* d_in, const int* in_sizes, int n_in,
                              void* d_out, int out_size, void* d_ws, size_t ws_size,
                              hipStream_t stream) {
  (void)in_sizes; (void)n_in; (void)out_size; (void)ws_size;

  const int cB = 2, cT = 2048, cE = 1024, cH = 8, cD = 128, cHID = 2730;
  const int cHIDp = 2752;                        // roundup32(2730)
  const int M = cB * cT;                         // 4096 rows
  const long long nBTE = (long long)M * cE;
  const long long nBTH = (long long)M * cHID;
  const long long nBTHp = (long long)M * cHIDp;

  const float* sdr = (const float*)d_in[0];
  const float* wq  = (const float*)d_in[1];
  const float* wk  = (const float*)d_in[2];
  const float* wv  = (const float*)d_in[3];
  const float* wo  = (const float*)d_in[4];
  const float* g1  = (const float*)d_in[5];
  const float* b1  = (const float*)d_in[6];
  const float* g2  = (const float*)d_in[7];
  const float* b2  = (const float*)d_in[8];
  const float* wg  = (const float*)d_in[9];
  const float* wu  = (const float*)d_in[10];
  const float* wd  = (const float*)d_in[11];
  float* out = (float*)d_out;

  char* ws = (char*)d_ws;
  size_t off = 0;
  auto take = [&](size_t bytes) -> char* {
    char* p = ws + off;
    off += (bytes + 255) & ~(size_t)255;
    return p;
  };
  unsigned short* wqT = (unsigned short*)take((size_t)cE * cE * 2);
  unsigned short* wkT = (unsigned short*)take((size_t)cE * cE * 2);
  unsigned short* wvT = (unsigned short*)take((size_t)cE * cE * 2);
  unsigned short* woT = (unsigned short*)take((size_t)cE * cE * 2);
  unsigned short* wgT = (unsigned short*)take((size_t)cHID * cE * 2);   // [2730][1024]
  unsigned short* wuT = (unsigned short*)take((size_t)cHID * cE * 2);   // [2730][1024]
  unsigned short* wdT = (unsigned short*)take((size_t)cE * cHIDp * 2);  // [1024][2752]
  unsigned short* xn_b = (unsigned short*)take((size_t)nBTE * 2);
  unsigned short* q_b  = (unsigned short*)take((size_t)nBTE * 2);
  unsigned short* k_b  = (unsigned short*)take((size_t)nBTE * 2);
  unsigned short* v_b  = (unsigned short*)take((size_t)nBTE * 2);
  float*          ctx  = (float*)take((size_t)nBTE * 4);
  float*          tmp32 = (float*)take((size_t)nBTE * 4);
  float*          x1    = (float*)take((size_t)nBTE * 4);
  unsigned short* gate_b = (unsigned short*)take((size_t)nBTH * 2);
  unsigned short* up_b   = (unsigned short*)take((size_t)nBTH * 2);
  unsigned short* h_b    = (unsigned short*)take((size_t)nBTHp * 2);
  unsigned short* vst_b  = q_b;   // q no longer needed after attention

  // 1) weights -> bf16, transposed, K-padded
  cvt_transpose_bf16<<<dim3(32, 32), 256, 0, stream>>>(wq, wqT, cE, cE, cE);
  cvt_transpose_bf16<<<dim3(32, 32), 256, 0, stream>>>(wk, wkT, cE, cE, cE);
  cvt_transpose_bf16<<<dim3(32, 32), 256, 0, stream>>>(wv, wvT, cE, cE, cE);
  cvt_transpose_bf16<<<dim3(32, 32), 256, 0, stream>>>(wo, woT, cE, cE, cE);
  cvt_transpose_bf16<<<dim3(86, 32), 256, 0, stream>>>(wg, wgT, cE, cHID, cE);
  cvt_transpose_bf16<<<dim3(86, 32), 256, 0, stream>>>(wu, wuT, cE, cHID, cE);
  cvt_transpose_bf16<<<dim3(32, 86), 256, 0, stream>>>(wd, wdT, cHID, cE, cHIDp);

  // 2) LN1 -> bf16
  ln_bf16<<<M, 256, 0, stream>>>(sdr, g1, b1, xn_b, cE);

  // 3) Q/K/V projections (bf16 out)
  {
    dim3 g(cE / 128, M / 64);
    gemm_bf16_wmma<<<g, 256, 0, stream>>>(xn_b, wqT, q_b, M, cE, cE, cE, cE, 1);
    gemm_bf16_wmma<<<g, 256, 0, stream>>>(xn_b, wkT, k_b, M, cE, cE, cE, cE, 1);
    gemm_bf16_wmma<<<g, 256, 0, stream>>>(xn_b, wvT, v_b, M, cE, cE, cE, cE, 1);
  }

  // 4) attention -> ctx (f32)
  attn_wmma<<<dim3(cT / 64, cB * cH), 128, 0, stream>>>(q_b, k_b, v_b, ctx,
                                                        cT, cE, cH, cD);

  // 5) FHN dynamics -> vst (bf16, overwrites q_b)
  fhn_bf16<<<blks(nBTE), 256, 0, stream>>>(ctx, vst_b, nBTE);

  // 6) O projection (f32 out), 7) residual x1 = sdr + attn_out
  gemm_bf16_wmma<<<dim3(cE / 128, M / 64), 256, 0, stream>>>(
      vst_b, woT, tmp32, M, cE, cE, cE, cE, 0);
  add_f32<<<blks(nBTE), 256, 0, stream>>>(sdr, tmp32, x1, nBTE);

  // 8) LN2 -> bf16
  ln_bf16<<<M, 256, 0, stream>>>(x1, g2, b2, xn_b, cE);

  // 9/10) gate & up projections (bf16 out, ragged N)
  {
    dim3 g((cHID + 127) / 128, M / 64);
    gemm_bf16_wmma<<<g, 256, 0, stream>>>(xn_b, wgT, gate_b, M, cHID, cE, cE, cE, 1);
    gemm_bf16_wmma<<<g, 256, 0, stream>>>(xn_b, wuT, up_b,  M, cHID, cE, cE, cE, 1);
  }

  // 11) h = silu(gate) * up -> zero-padded [M][2752]
  silu_mul_pad_bf16<<<blks(nBTHp), 256, 0, stream>>>(gate_b, up_b, h_b,
                                                     cHID, cHIDp, nBTHp);

  // 12) down projection (f32 out, K fully padded), 13) final residual
  gemm_bf16_wmma<<<dim3(cE / 128, M / 64), 256, 0, stream>>>(
      h_b, wdT, tmp32, M, cE, cHID, cHIDp, cHIDp, 0);
  add_f32<<<blks(nBTE), 256, 0, stream>>>(x1, tmp32, out, nBTE);
}
